// MoE_38628935860429
// MI455X (gfx1250) — matrix operand order repaired and checked by reading, not compiled, not run on hardware
//
#include <hip/hip_runtime.h>

#define NTOK 8192
#define CDIM 1024
#define HDIM 2048
#define ENUM 8

typedef __attribute__((ext_vector_type(16))) __bf16 v16bf;
typedef __attribute__((ext_vector_type(8)))  __bf16 bf16x8;
typedef __attribute__((ext_vector_type(2)))  __bf16 bf16x2;
typedef __attribute__((ext_vector_type(8)))  float  v8f;

union FragBf { v16bf v; bf16x8 h[2]; };

static __device__ inline unsigned pack2(float a, float b) {
  union { bf16x2 v; unsigned u; } t;
  t.v.x = (__bf16)a; t.v.y = (__bf16)b;
  return t.u;
}

static __device__ inline v8f vzero8() {
  v8f z;
#pragma unroll
  for (int i = 0; i < 8; ++i) z[i] = 0.0f;
  return z;
}

static __device__ inline void atomic_add_f32(float* p, float v) {
  __hip_atomic_fetch_add(p, v, __ATOMIC_RELAXED, __HIP_MEMORY_SCOPE_AGENT);
}

static __device__ inline float fast_silu(float g) {
  // g * sigmoid(g) with raw v_rcp_f32 (result is truncated to bf16 anyway)
  return g * __builtin_amdgcn_rcpf(1.0f + __expf(-g));
}

// ---------------------------------------------------------------- init ----
__global__ __launch_bounds__(256) void moe_init_kernel(float* __restrict__ out,
                                                       int* __restrict__ counts) {
  size_t i = (size_t)blockIdx.x * 256 + threadIdx.x;   // 4 floats each
  float4 z; z.x = 0.f; z.y = 0.f; z.z = 0.f; z.w = 0.f;
  ((float4*)out)[i] = z;
  if (blockIdx.x == 0 && threadIdx.x < ENUM) counts[threadIdx.x] = 0;
}

// ------------------------------------------------------------- cvt x -----
__global__ __launch_bounds__(256) void moe_cvtx_kernel(const float* __restrict__ x,
                                                       __bf16* __restrict__ xb) {
  size_t i = (size_t)blockIdx.x * 256 + threadIdx.x;   // 8 floats each
  const float4* xp = (const float4*)x + i * 2;
  float4 a = xp[0], b = xp[1];
  bf16x8 o;
  o[0] = (__bf16)a.x; o[1] = (__bf16)a.y; o[2] = (__bf16)a.z; o[3] = (__bf16)a.w;
  o[4] = (__bf16)b.x; o[5] = (__bf16)b.y; o[6] = (__bf16)b.z; o[7] = (__bf16)b.w;
  ((bf16x8*)xb)[i] = o;
}

// ------------------------------------------------------------- gating ----
__global__ __launch_bounds__(256) void moe_gate_kernel(const float* __restrict__ x,
                                                       const float* __restrict__ wgt,
                                                       int* __restrict__ list,
                                                       int* __restrict__ counts,
                                                       float* __restrict__ cw) {
  const int lane = threadIdx.x & 31;
  const int n = blockIdx.x * 8 + (threadIdx.x >> 5);
  float acc[8];
#pragma unroll
  for (int e = 0; e < 8; ++e) acc[e] = 0.f;
  const float* xr = x + (size_t)n * CDIM;
  for (int c = lane; c < CDIM; c += 32) {
    float xv = xr[c];
    float4 g0 = *(const float4*)(wgt + c * 8);
    float4 g1 = *(const float4*)(wgt + c * 8 + 4);
    acc[0] += xv * g0.x; acc[1] += xv * g0.y; acc[2] += xv * g0.z; acc[3] += xv * g0.w;
    acc[4] += xv * g1.x; acc[5] += xv * g1.y; acc[6] += xv * g1.z; acc[7] += xv * g1.w;
  }
#pragma unroll
  for (int e = 0; e < 8; ++e) {
#pragma unroll
    for (int off = 16; off > 0; off >>= 1)
      acc[e] += __shfl_xor(acc[e], off, 32);
  }
  if (lane == 0) {
    float m = acc[0];
#pragma unroll
    for (int e = 1; e < 8; ++e) m = fmaxf(m, acc[e]);
    float g[8], s = 0.f;
#pragma unroll
    for (int e = 0; e < 8; ++e) { g[e] = __expf(acc[e] - m); s += g[e]; }
    float inv = 1.0f / s;
#pragma unroll
    for (int e = 0; e < 8; ++e) g[e] *= inv;
    int e1 = 0; float v1 = g[0];
#pragma unroll
    for (int e = 1; e < 8; ++e) if (g[e] > v1) { v1 = g[e]; e1 = e; }
    int e2 = -1; float v2 = -1.0f;
#pragma unroll
    for (int e = 0; e < 8; ++e) if (e != e1 && g[e] > v2) { v2 = g[e]; e2 = e; }
    cw[n * 2 + 0] = v1;
    cw[n * 2 + 1] = v2;
    int p = atomicAdd(&counts[e1], 1);
    list[e1 * NTOK + p] = n;                 // slot 0
    int q = atomicAdd(&counts[e2], 1);
    list[e2 * NTOK + q] = n | 0x10000;       // slot 1
  }
}

// ----------------------------------------------------- gemm1: up/gate ----
// Block tile: 64 tokens x 128 H cols. 8 waves in a 2x4 grid of 32x32 subtiles.
// Double-buffered LDS: one barrier per K-step; next tile's global loads are
// issued before the WMMA block so VMEM latency overlaps matrix math.
__global__ __launch_bounds__(256) void moe_gemm1_kernel(
    const __bf16* __restrict__ xb, const float* __restrict__ wi,
    const float* __restrict__ wg, const int* __restrict__ list,
    const int* __restrict__ counts, __bf16* __restrict__ hbuf) {
  const int e = blockIdx.z;
  int cnt = counts[e]; if (cnt > NTOK) cnt = NTOK;
  const int m0 = blockIdx.y * 64;
  if (m0 >= cnt) return;
  const int h0 = blockIdx.x * 128;

  __shared__ __bf16   Ald[2][64 * 40];    // [row][k], row stride 40 bf16 (80B)
  __shared__ unsigned Bi[2][128 * 20];    // [col][kpair], col stride 20 dw (80B)
  __shared__ unsigned Bg[2][128 * 20];
  __shared__ int tokA[64];
  __shared__ int dstR[64];

  const int tid = threadIdx.x;
  if (tid < 64) {
    int idx = m0 + tid;
    if (idx < cnt) {
      int v = list[e * NTOK + idx];
      tokA[tid] = v & 0xFFFF;
      dstR[tid] = (v & 0xFFFF) * 2 + (v >> 16);
    } else { tokA[tid] = 0; dstR[tid] = -1; }
  }
  __syncthreads();

  const int w  = tid >> 5;
  const int ln = tid & 15;
  const int kh = (tid >> 4) & 1;
  const int wm = (w & 1) * 32;
  const int wn = (w >> 1) * 32;

  const int arow = tid >> 2, aseg = tid & 3;
  const int bc0 = (tid & 31) * 4, bkb = (tid >> 5) * 4;
  const size_t wbase = (size_t)e * CDIM * HDIM;
  const __bf16* aptr = xb + (size_t)tokA[arow] * CDIM + aseg * 8;
  const float*  bip  = wi + wbase + (size_t)bkb * HDIM + h0 + bc0;
  const float*  bgp  = wg + wbase + (size_t)bkb * HDIM + h0 + bc0;
  const int abase = arow * 40 + aseg * 8;
  const int bbase = bc0 * 20 + (bkb >> 1);   // +j*20 per col

  // ---- stage tile 0 into buffer 0
  {
    bf16x8 av = *(const bf16x8*)aptr;
    const float* bi = bip;
    const float* bg = bgp;
    float4 i0 = *(const float4*)bi;
    float4 i1 = *(const float4*)(bi + HDIM);
    float4 i2 = *(const float4*)(bi + 2 * HDIM);
    float4 i3 = *(const float4*)(bi + 3 * HDIM);
    float4 g0 = *(const float4*)bg;
    float4 g1 = *(const float4*)(bg + HDIM);
    float4 g2 = *(const float4*)(bg + 2 * HDIM);
    float4 g3 = *(const float4*)(bg + 3 * HDIM);
    *(bf16x8*)&Ald[0][abase] = av;
#pragma unroll
    for (int j = 0; j < 4; ++j) {
      Bi[0][bbase + j * 20]     = pack2((&i0.x)[j], (&i1.x)[j]);
      Bi[0][bbase + j * 20 + 1] = pack2((&i2.x)[j], (&i3.x)[j]);
      Bg[0][bbase + j * 20]     = pack2((&g0.x)[j], (&g1.x)[j]);
      Bg[0][bbase + j * 20 + 1] = pack2((&g2.x)[j], (&g3.x)[j]);
    }
  }
  __syncthreads();

  v8f aU[2][2], aG[2][2];
#pragma unroll
  for (int i = 0; i < 2; ++i)
#pragma unroll
    for (int j = 0; j < 2; ++j) { aU[i][j] = vzero8(); aG[i][j] = vzero8(); }

  int cur = 0;
  for (int kk = 0; kk < CDIM; kk += 32) {
    const int nxt = cur ^ 1;
    const bool pf = (kk + 32) < CDIM;

    // ---- issue next tile's global loads (overlap with WMMA below)
    bf16x8 av;
    float4 i0, i1, i2, i3, g0, g1, g2, g3;
    if (pf) {
      av = *(const bf16x8*)(aptr + kk + 32);
      const float* bi = bip + (size_t)(kk + 32) * HDIM;
      const float* bg = bgp + (size_t)(kk + 32) * HDIM;
      i0 = *(const float4*)bi;
      i1 = *(const float4*)(bi + HDIM);
      i2 = *(const float4*)(bi + 2 * HDIM);
      i3 = *(const float4*)(bi + 3 * HDIM);
      g0 = *(const float4*)bg;
      g1 = *(const float4*)(bg + HDIM);
      g2 = *(const float4*)(bg + 2 * HDIM);
      g3 = *(const float4*)(bg + 3 * HDIM);
    }

    // ---- compute from current buffer
    FragBf fa[2], fbi[2], fbg[2];
#pragma unroll
    for (int i = 0; i < 2; ++i) {
      int m = wm + i * 16 + ln;
      fa[i].h[0] = *(const bf16x8*)&Ald[cur][m * 40 + kh * 16];
      fa[i].h[1] = *(const bf16x8*)&Ald[cur][m * 40 + kh * 16 + 8];
    }
#pragma unroll
    for (int j = 0; j < 2; ++j) {
      int n = wn + j * 16 + ln;
      fbi[j].h[0] = *(const bf16x8*)&Bi[cur][n * 20 + kh * 8];
      fbi[j].h[1] = *(const bf16x8*)&Bi[cur][n * 20 + kh * 8 + 4];
      fbg[j].h[0] = *(const bf16x8*)&Bg[cur][n * 20 + kh * 8];
      fbg[j].h[1] = *(const bf16x8*)&Bg[cur][n * 20 + kh * 8 + 4];
    }
#pragma unroll
    for (int i = 0; i < 2; ++i)
#pragma unroll
      for (int j = 0; j < 2; ++j) {
        aU[i][j] = __builtin_amdgcn_wmma_f32_16x16x32_bf16(
            false, fa[i].v, false, fbi[j].v, (short)0, aU[i][j], false, false);
        aG[i][j] = __builtin_amdgcn_wmma_f32_16x16x32_bf16(
            false, fa[i].v, false, fbg[j].v, (short)0, aG[i][j], false, false);
      }

    // ---- commit next tile to the other buffer
    if (pf) {
      *(bf16x8*)&Ald[nxt][abase] = av;
#pragma unroll
      for (int j = 0; j < 4; ++j) {
        Bi[nxt][bbase + j * 20]     = pack2((&i0.x)[j], (&i1.x)[j]);
        Bi[nxt][bbase + j * 20 + 1] = pack2((&i2.x)[j], (&i3.x)[j]);
        Bg[nxt][bbase + j * 20]     = pack2((&g0.x)[j], (&g1.x)[j]);
        Bg[nxt][bbase + j * 20 + 1] = pack2((&g2.x)[j], (&g3.x)[j]);
      }
    }
    __syncthreads();
    cur = nxt;
  }

  // epilogue: h = silu(gate) * up, store bf16 at slot-row
#pragma unroll
  for (int i = 0; i < 2; ++i) {
#pragma unroll
    for (int v = 0; v < 8; ++v) {
      int mloc = wm + i * 16 + kh * 8 + v;
      int r = dstR[mloc];
      if (r < 0) continue;
#pragma unroll
      for (int j = 0; j < 2; ++j) {
        float hv = aU[i][j][v] * fast_silu(aG[i][j][v]);
        hbuf[(size_t)r * HDIM + h0 + wn + j * 16 + ln] = (__bf16)hv;
      }
    }
  }
}

// ------------------------------------------------ gemm2: h @ wo -> out ---
__global__ __launch_bounds__(256) void moe_gemm2_kernel(
    const __bf16* __restrict__ hbuf, const float* __restrict__ wo,
    const int* __restrict__ list, const int* __restrict__ counts,
    const float* __restrict__ cw, float* __restrict__ out) {
  const int e = blockIdx.z;
  int cnt = counts[e]; if (cnt > NTOK) cnt = NTOK;
  const int m0 = blockIdx.y * 64;
  if (m0 >= cnt) return;
  const int c0o = blockIdx.x * 128;

  __shared__ __bf16   Ald[2][64 * 40];
  __shared__ unsigned Bo[2][128 * 20];
  __shared__ int   rowA[64];
  __shared__ int   tokT[64];
  __shared__ float cwv[64];

  const int tid = threadIdx.x;
  if (tid < 64) {
    int idx = m0 + tid;
    if (idx < cnt) {
      int v = list[e * NTOK + idx];
      int tok = v & 0xFFFF;
      int r = tok * 2 + (v >> 16);
      rowA[tid] = r; tokT[tid] = tok; cwv[tid] = cw[r];
    } else { rowA[tid] = 0; tokT[tid] = -1; cwv[tid] = 0.f; }
  }
  __syncthreads();

  const int w  = tid >> 5;
  const int ln = tid & 15;
  const int kh = (tid >> 4) & 1;
  const int wm = (w & 1) * 32;
  const int wn = (w >> 1) * 32;

  const int arow = tid >> 2, aseg = tid & 3;
  const int bc0 = (tid & 31) * 4, bkb = (tid >> 5) * 4;
  const size_t wbase = (size_t)e * HDIM * CDIM;
  const __bf16* aptr = hbuf + (size_t)rowA[arow] * HDIM + aseg * 8;
  const float*  bop  = wo + wbase + (size_t)bkb * CDIM + c0o + bc0;
  const int abase = arow * 40 + aseg * 8;
  const int bbase = bc0 * 20 + (bkb >> 1);

  // ---- stage tile 0
  {
    bf16x8 av = *(const bf16x8*)aptr;
    float4 o0 = *(const float4*)bop;
    float4 o1 = *(const float4*)(bop + CDIM);
    float4 o2 = *(const float4*)(bop + 2 * CDIM);
    float4 o3 = *(const float4*)(bop + 3 * CDIM);
    *(bf16x8*)&Ald[0][abase] = av;
#pragma unroll
    for (int j = 0; j < 4; ++j) {
      Bo[0][bbase + j * 20]     = pack2((&o0.x)[j], (&o1.x)[j]);
      Bo[0][bbase + j * 20 + 1] = pack2((&o2.x)[j], (&o3.x)[j]);
    }
  }
  __syncthreads();

  v8f acc[2][2];
#pragma unroll
  for (int i = 0; i < 2; ++i)
#pragma unroll
    for (int j = 0; j < 2; ++j) acc[i][j] = vzero8();

  int cur = 0;
  for (int kk = 0; kk < HDIM; kk += 32) {
    const int nxt = cur ^ 1;
    const bool pf = (kk + 32) < HDIM;

    bf16x8 av;
    float4 o0, o1, o2, o3;
    if (pf) {
      av = *(const bf16x8*)(aptr + kk + 32);
      const float* bo = bop + (size_t)(kk + 32) * CDIM;
      o0 = *(const float4*)bo;
      o1 = *(const float4*)(bo + CDIM);
      o2 = *(const float4*)(bo + 2 * CDIM);
      o3 = *(const float4*)(bo + 3 * CDIM);
    }

    FragBf fa[2], fb[2];
#pragma unroll
    for (int i = 0; i < 2; ++i) {
      int m = wm + i * 16 + ln;
      fa[i].h[0] = *(const bf16x8*)&Ald[cur][m * 40 + kh * 16];
      fa[i].h[1] = *(const bf16x8*)&Ald[cur][m * 40 + kh * 16 + 8];
    }
#pragma unroll
    for (int j = 0; j < 2; ++j) {
      int n = wn + j * 16 + ln;
      fb[j].h[0] = *(const bf16x8*)&Bo[cur][n * 20 + kh * 8];
      fb[j].h[1] = *(const bf16x8*)&Bo[cur][n * 20 + kh * 8 + 4];
    }
#pragma unroll
    for (int i = 0; i < 2; ++i)
#pragma unroll
      for (int j = 0; j < 2; ++j)
        acc[i][j] = __builtin_amdgcn_wmma_f32_16x16x32_bf16(
            false, fa[i].v, false, fb[j].v, (short)0, acc[i][j], false, false);

    if (pf) {
      *(bf16x8*)&Ald[nxt][abase] = av;
#pragma unroll
      for (int j = 0; j < 4; ++j) {
        Bo[nxt][bbase + j * 20]     = pack2((&o0.x)[j], (&o1.x)[j]);
        Bo[nxt][bbase + j * 20 + 1] = pack2((&o2.x)[j], (&o3.x)[j]);
      }
    }
    __syncthreads();
    cur = nxt;
  }

  // epilogue: out[token] += combine * y  (two commutative fp adds per elem)
#pragma unroll
  for (int i = 0; i < 2; ++i) {
#pragma unroll
    for (int v = 0; v < 8; ++v) {
      int mloc = wm + i * 16 + kh * 8 + v;
      int tok = tokT[mloc];
      if (tok < 0) continue;
      float wgt = cwv[mloc];
#pragma unroll
      for (int j = 0; j < 2; ++j) {
        int col = c0o + wn + j * 16 + ln;
        atomic_add_f32(&out[(size_t)tok * CDIM + col], wgt * acc[i][j][v]);
      }
    }
  }
}

// ---------------------------------------------------------------- host ---
extern "C" void kernel_launch(void* const* d_in, const int* in_sizes, int n_in,
                              void* d_out, int out_size, void* d_ws, size_t ws_size,
                              hipStream_t stream) {
  (void)in_sizes; (void)n_in; (void)out_size; (void)ws_size;
  const float* x      = (const float*)d_in[0];
  const float* w_gate = (const float*)d_in[1];
  const float* wi     = (const float*)d_in[2];
  const float* wg     = (const float*)d_in[3];
  const float* wo     = (const float*)d_in[4];
  float* out = (float*)d_out;

  char* ws = (char*)d_ws;
  __bf16* xb   = (__bf16*)ws;                                         // 16 MB
  __bf16* hbuf = (__bf16*)(ws + (size_t)NTOK * CDIM * 2);             // 64 MB
  int*    list = (int*)(ws + (size_t)NTOK * CDIM * 2 + (size_t)2 * NTOK * HDIM * 2);
  float*  cw   = (float*)((char*)list + (size_t)ENUM * NTOK * 4);
  int*  counts = (int*)((char*)cw + (size_t)2 * NTOK * 4);

  moe_init_kernel<<<dim3((NTOK * CDIM) / (4 * 256)), 256, 0, stream>>>(out, counts);
  moe_cvtx_kernel<<<dim3((NTOK * CDIM) / (8 * 256)), 256, 0, stream>>>(x, xb);
  moe_gate_kernel<<<dim3(NTOK / 8), 256, 0, stream>>>(x, w_gate, list, counts, cw);
  moe_gemm1_kernel<<<dim3(HDIM / 128, NTOK / 64, ENUM), 256, 0, stream>>>(
      xb, wi, wg, list, counts, hbuf);
  moe_gemm2_kernel<<<dim3(CDIM / 128, NTOK / 64, ENUM), 256, 0, stream>>>(
      hbuf, wo, list, counts, cw, out);
}
